// SimpleSelfAttention_87376814670069
// MI455X (gfx1250) — matrix operand order repaired and checked by reading, not compiled, not run on hardware
//
#include <hip/hip_runtime.h>
#include <hip/hip_bf16.h>

typedef __attribute__((ext_vector_type(16))) __bf16 v16bf;
typedef __attribute__((ext_vector_type(8)))  float  v8f;

#define Bc 16
#define Tc 4096
#define Dc 512
#define D2 1024

static __device__ __forceinline__ unsigned pack_bf16(float a, float b) {
  unsigned ua = __float_as_uint(a), ub = __float_as_uint(b);
  ua = (ua + 0x7FFFu + ((ua >> 16) & 1u)) >> 16;   // round-to-nearest-even
  ub = (ub + 0x7FFFu + ((ub >> 16) & 1u)) >> 16;
  return (ub << 16) | (ua & 0xFFFFu);
}

union BU { uint4 q[2]; v16bf v; };

// --- K0a: swizzle W1 (fp32 [1024,512]) into bf16 WMMA B-matrix layout ------
// B = W1^T (K=512 x N=1024), tiled 16 N-cols x 32 K per WMMA step.
// dword idx = ((jt*16+ks)*32 + lane)*8 + p ; lane = half*16 + (j%16),
// k = ks*32 + half*8 + 2(p&3) + 16(p>>2) + q  (q = lo/hi half-word)
__global__ void k_prep_b(const float* __restrict__ W1, unsigned* __restrict__ Bsw) {
  int idx  = blockIdx.x * blockDim.x + threadIdx.x;   // 262144 dwords
  int p    = idx & 7;
  int lane = (idx >> 3) & 31;
  int ks   = (idx >> 8) & 15;
  int jt   = idx >> 12;
  int j    = jt * 16 + (lane & 15);
  int k    = ks * 32 + (lane >> 4) * 8 + 2 * (p & 3) + 16 * (p >> 2);
  const float2 f = *(const float2*)(W1 + (size_t)j * Dc + k);
  Bsw[idx] = pack_bf16(f.x, f.y);
}

// --- K0b: c[b,j] = hn[b]·W2[j,:] + W1_b[j] + W2_b[j] ----------------------
__global__ void k_prep_c(const float* __restrict__ hn, const float* __restrict__ W2,
                         const float* __restrict__ W1b, const float* __restrict__ W2b,
                         float* __restrict__ cvec) {
  int lane = threadIdx.x & 31, wid = threadIdx.x >> 5;
  int Wv = blockIdx.x * (blockDim.x >> 5) + wid;      // 16384 waves
  int b = Wv >> 10, j = Wv & 1023;
  float acc = 0.f;
  for (int k = lane; k < Dc; k += 32) acc += hn[b * Dc + k] * W2[(size_t)j * Dc + k];
  for (int m = 16; m >= 1; m >>= 1) acc += __shfl_xor(acc, m, 32);
  if (lane == 0) cvec[b * D2 + j] = acc + W1b[j] + W2b[j];
}

// --- K1: fused scores[b,t] = V · tanh(h W1^T + c_b) + V_b  (WMMA bf16) ----
// Each wave owns TWO 16-row M-tiles (32 rows; A tiles are staged in LDS,
// re-read per k-step). B tiles are software-pipelined 2 k-steps ahead through a
// 3-deep rotating register set so L2 latency is hidden behind WMMAs.
__global__ void __launch_bounds__(128)
k_score(const float* __restrict__ h, const unsigned* __restrict__ Bsw,
        const float* __restrict__ cvec, const float* __restrict__ Vw,
        const float* __restrict__ Vb, float* __restrict__ scores) {
  extern __shared__ unsigned alds[];                  // 128KB: 4 waves * 2 tiles * 16KB
  int lane = threadIdx.x & 31, wid = threadIdx.x >> 5;
  int gw = blockIdx.x * 4 + wid;                      // 2048 waves, 32 rows each
  int b  = gw >> 7;                                   // 128 waves per batch
  int t0 = (gw & 127) << 5;
  unsigned* my = alds + wid * 8192;                   // 2 x 4096 dwords

  // stage 2 x (16x512) fp32 -> bf16 in WMMA A-layout (ISA 7.12.2, 16-bit A 16x32)
  for (int u = 0; u < 2; ++u) {
    const float* hblk = h + ((size_t)(b * Tc + t0 + u * 16)) * Dc;
    unsigned* dstbuf = my + u * 4096;
    for (int i = 0; i < 64; ++i) {
      int qi = i * 32 + lane;               // quad of 4 consecutive floats
      int e  = qi * 4;
      int row = e >> 9;
      int kf  = e & 511;
      const float4 f = *(const float4*)(hblk + (size_t)e);
      int ks   = kf >> 5;
      int kk   = kf & 31;
      int half = (kk >> 3) & 1;
      int p    = ((kk & 16) ? 4 : 0) + ((kk >> 1) & 3);
      int lp   = half * 16 + row;
      unsigned* dst = &dstbuf[(ks * 32 + lp) * 8 + p];
      dst[0] = pack_bf16(f.x, f.y);
      dst[1] = pack_bf16(f.z, f.w);
    }
  }
  __syncthreads();

  int laneN = lane & 15;
  float acc0[8] = {0.f,0.f,0.f,0.f,0.f,0.f,0.f,0.f};
  float acc1[8] = {0.f,0.f,0.f,0.f,0.f,0.f,0.f,0.f};
  const uint4* B4 = (const uint4*)Bsw;

  // prime 2-deep B pipeline (linear step = jt*16+ks, 1024 steps total)
  BU Bcur, Bn1;
  {
    int bi0 = (0 * 32 + lane) * 2;
    Bcur.q[0] = B4[bi0]; Bcur.q[1] = B4[bi0 + 1];
    int bi1 = (1 * 32 + lane) * 2;
    Bn1.q[0] = B4[bi1]; Bn1.q[1] = B4[bi1 + 1];
  }

  for (int jt = 0; jt < 64; ++jt) {
    v8f C0 = {0.f,0.f,0.f,0.f,0.f,0.f,0.f,0.f};
    v8f C1 = {0.f,0.f,0.f,0.f,0.f,0.f,0.f,0.f};
    #pragma unroll
    for (int ks = 0; ks < 16; ++ks) {
      // prefetch B tile 2 steps ahead (clamped at the last tile)
      int nstep = jt * 16 + ks + 2;
      if (nstep > 1023) nstep = 1023;
      BU Bn2;
      int bi = (nstep * 32 + lane) * 2;
      Bn2.q[0] = B4[bi]; Bn2.q[1] = B4[bi + 1];

      union { uint4 q[2]; v16bf v; } A0, A1;
      const uint4* a0 = (const uint4*)&my[(ks * 32 + lane) * 8];
      A0.q[0] = a0[0]; A0.q[1] = a0[1];
      const uint4* a1 = (const uint4*)&my[4096 + (ks * 32 + lane) * 8];
      A1.q[0] = a1[0]; A1.q[1] = a1[1];

      C0 = __builtin_amdgcn_wmma_f32_16x16x32_bf16(false, A0.v, false, Bcur.v,
                                                   (short)0, C0, false, false);
      C1 = __builtin_amdgcn_wmma_f32_16x16x32_bf16(false, A1.v, false, Bcur.v,
                                                   (short)0, C1, false, false);
      Bcur = Bn1;
      Bn1  = Bn2;
    }
    int j = jt * 16 + laneN;
    float cj = cvec[b * D2 + j];
    float vj = Vw[j];
    #pragma unroll
    for (int r = 0; r < 8; ++r) {
      acc0[r] += vj * tanhf(C0[r] + cj);
      acc1[r] += vj * tanhf(C1[r] + cj);
    }
  }
  // reduce the 16 N-lanes in each half (rows r+8*half live in lane halves)
  #pragma unroll
  for (int r = 0; r < 8; ++r) {
    float v0 = acc0[r], v1 = acc1[r];
    v0 += __shfl_xor(v0, 1, 32);  v1 += __shfl_xor(v1, 1, 32);
    v0 += __shfl_xor(v0, 2, 32);  v1 += __shfl_xor(v1, 2, 32);
    v0 += __shfl_xor(v0, 4, 32);  v1 += __shfl_xor(v1, 4, 32);
    v0 += __shfl_xor(v0, 8, 32);  v1 += __shfl_xor(v1, 8, 32);
    acc0[r] = v0; acc1[r] = v1;
  }
  if (laneN == 0) {
    int half = lane >> 4;
    float vb = Vb[0];
    #pragma unroll
    for (int r = 0; r < 8; ++r) {
      scores[b * Tc + t0 +      half * 8 + r] = acc0[r] + vb;
      scores[b * Tc + t0 + 16 + half * 8 + r] = acc1[r] + vb;
    }
  }
}

// --- K2: masked softmax over T, in place -----------------------------------
__global__ void __launch_bounds__(256)
k_softmax(const unsigned char* __restrict__ mask, float* __restrict__ scores) {
  __shared__ float red[8];
  int b = blockIdx.x, tid = threadIdx.x;
  int lane = tid & 31, wid = tid >> 5;
  float loc[16];
  float mx = -3.4e38f;
  for (int i = 0; i < 16; ++i) {
    int t = i * 256 + tid;
    float s = scores[b * Tc + t];
    if (mask[b * Tc + t]) s = -1e12f;
    loc[i] = s;
    mx = fmaxf(mx, s);
  }
  for (int m = 16; m >= 1; m >>= 1) mx = fmaxf(mx, __shfl_xor(mx, m, 32));
  if (lane == 0) red[wid] = mx;
  __syncthreads();
  float gmx = red[0];
  for (int w = 1; w < 8; ++w) gmx = fmaxf(gmx, red[w]);
  __syncthreads();
  float sum = 0.f;
  for (int i = 0; i < 16; ++i) { float e = __expf(loc[i] - gmx); loc[i] = e; sum += e; }
  for (int m = 16; m >= 1; m >>= 1) sum += __shfl_xor(sum, m, 32);
  if (lane == 0) red[wid] = sum;
  __syncthreads();
  float gs = 0.f;
  for (int w = 0; w < 8; ++w) gs += red[w];
  float inv = 1.0f / gs;
  for (int i = 0; i < 16; ++i) scores[b * Tc + i * 256 + tid] = loc[i] * inv;
}

// --- K3: ctx partials: sum over a T-chunk of attn[b,t]*h[b,t,d] ------------
__global__ void __launch_bounds__(256)
k_ctx_partial(const float* __restrict__ h, const float* __restrict__ attn,
              float* __restrict__ ctxp) {
  int id = blockIdx.x;                 // 16 * 2 * 32 = 1024 blocks
  int tc = id & 31, dblk = (id >> 5) & 1, b = id >> 6;
  int d = dblk * 256 + threadIdx.x;
  const float* hb = h + ((size_t)b * Tc) * Dc;
  const float* ab = attn + b * Tc;
  float acc = 0.f;
  for (int tt = 0; tt < 128; ++tt) {
    int t = tc * 128 + tt;
    acc += ab[t] * hb[(size_t)t * Dc + d];
  }
  ctxp[(b * 32 + tc) * Dc + d] = acc;
}

__global__ void k_ctx_reduce(const float* __restrict__ ctxp, float* __restrict__ ctx) {
  int idx = blockIdx.x * blockDim.x + threadIdx.x;   // 8192
  int b = idx >> 9, d = idx & 511;
  float acc = 0.f;
  for (int tc = 0; tc < 32; ++tc) acc += ctxp[(b * 32 + tc) * Dc + d];
  ctx[idx] = acc;
}

// --- K4: out[b,i] = Wo[i,0:512]·ctx[b] + Wo[i,512:1024]·hn[b] + Wo_b[i] ----
__global__ void __launch_bounds__(256)
k_out(const float* __restrict__ ctx, const float* __restrict__ hn,
      const float* __restrict__ Wo, const float* __restrict__ Wob,
      float* __restrict__ out) {
  int lane = threadIdx.x & 31, wid = threadIdx.x >> 5;
  int Wv = blockIdx.x * 8 + wid;                     // 8192 waves
  int b = Wv >> 9, i = Wv & 511;
  const float* wrow = Wo + (size_t)i * D2;
  float acc = 0.f;
  for (int d = lane; d < Dc; d += 32)
    acc += ctx[b * Dc + d] * wrow[d] + hn[b * Dc + d] * wrow[Dc + d];
  for (int m = 16; m >= 1; m >>= 1) acc += __shfl_xor(acc, m, 32);
  if (lane == 0) out[b * Dc + i] = acc + Wob[i];
}

extern "C" void kernel_launch(void* const* d_in, const int* in_sizes, int n_in,
                              void* d_out, int out_size, void* d_ws, size_t ws_size,
                              hipStream_t stream) {
  (void)in_sizes; (void)n_in; (void)out_size; (void)ws_size;
  const float*         h    = (const float*)d_in[0];
  const unsigned char* mask = (const unsigned char*)d_in[1];
  const float*         hn   = (const float*)d_in[2];
  const float*         W1w  = (const float*)d_in[3];
  const float*         W1b  = (const float*)d_in[4];
  const float*         W2w  = (const float*)d_in[5];
  const float*         W2b  = (const float*)d_in[6];
  const float*         Vw   = (const float*)d_in[7];
  const float*         Vb   = (const float*)d_in[8];
  const float*         Wow  = (const float*)d_in[9];
  const float*         Wob  = (const float*)d_in[10];
  float* out = (float*)d_out;

  char* ws = (char*)d_ws;
  unsigned* Bsw    = (unsigned*)(ws);                                   // 1 MiB
  float*    cvec   = (float*)(ws + 1048576);                            // 64 KiB
  float*    scores = (float*)(ws + 1048576 + 65536);                    // 256 KiB
  float*    ctxp   = (float*)(ws + 1048576 + 65536 + 262144);           // 1 MiB
  float*    ctx    = (float*)(ws + 1048576 + 65536 + 262144 + 1048576); // 32 KiB

  k_prep_b     <<<1024, 256, 0, stream>>>(W1w, Bsw);
  k_prep_c     <<<2048, 256, 0, stream>>>(hn, W2w, W1b, W2b, cvec);
  k_score      <<<512,  128, 131072, stream>>>(h, Bsw, cvec, Vw, Vb, scores);
  k_softmax    <<<16,   256, 0, stream>>>(mask, scores);
  k_ctx_partial<<<1024, 256, 0, stream>>>(h, scores, ctxp);
  k_ctx_reduce <<<32,   256, 0, stream>>>(ctxp, ctx);
  k_out        <<<1024, 256, 0, stream>>>(ctx, hn, Wow, Wob, out);
}
